// TauTransformer_78314433675751
// MI455X (gfx1250) — matrix operand order repaired and checked by reading, not compiled, not run on hardware
//
#include <hip/hip_runtime.h>
#include <math.h>

#define BDIM 256
#define TILE 16
#define NWAVE 8

typedef __attribute__((ext_vector_type(16))) _Float16 v16h;
typedef __attribute__((ext_vector_type(8)))  float    v8f;
typedef unsigned short u16;
typedef unsigned int   u32;

// ---------- f32 <-> f16 helpers ----------
__device__ __forceinline__ u16 f2h(float f) {
  union { _Float16 h; u16 u; } c; c.h = (_Float16)f; return c.u;
}
__device__ __forceinline__ float h2f(u16 u) {
  union { u16 u; _Float16 h; } c; c.u = u; return (float)c.h;
}
__device__ __forceinline__ float n2n(float v) {          // jnp.nan_to_num
  if (v != v) return 0.f;
  if (v >  3.4028235e38f) return  3.4028235e38f;
  if (v < -3.4028235e38f) return -3.4028235e38f;
  return v;
}

// ---------- WMMA fragment loaders (CDNA5 16x16x32 f16 layouts) ----------
__device__ __forceinline__ v16h frag8(const u16* p0, const u16* p1) {
  union { v16h v; u32 u[8]; } f;
  const u32* a = (const u32*)p0;
  const u32* b = (const u32*)p1;
  f.u[0]=a[0]; f.u[1]=a[1]; f.u[2]=a[2]; f.u[3]=a[3];
  f.u[4]=b[0]; f.u[5]=b[1]; f.u[6]=b[2]; f.u[7]=b[3];
  return f.v;
}
// A: row-major M x K (row stride lda). Lane l: M = m0+(l&15), K-half = (l>>4)*8.
__device__ __forceinline__ v16h loadA(const u16* A, int lda, int m0, int k0) {
  int lane = (int)(threadIdx.x & 31u);
  const u16* base = A + (size_t)(m0 + (lane & 15)) * lda + (k0 + ((lane >> 4) << 3));
  return frag8(base, base + 16);
}
// B stored transposed Bt[N][K] (row stride ldb). Lane l: N = n0+(l&15), K-half = (l>>4)*8.
__device__ __forceinline__ v16h loadB(const u16* Bt, int ldb, int n0, int k0) {
  int lane = (int)(threadIdx.x & 31u);
  const u16* base = Bt + (size_t)(n0 + (lane & 15)) * ldb + (k0 + ((lane >> 4) << 3));
  return frag8(base, base + 16);
}

// Double-buffered K loop: loads for step k+32 issued before WMMA of step k so the
// compiler can use a relaxed s_wait_dscnt and overlap LDS traffic with the matrix pipe.
__device__ __forceinline__ v8f kloop(const u16* A, int lda, const u16* Bt, int ldb,
                                     int K, int m0, int n0) {
  v8f acc = {0.f,0.f,0.f,0.f,0.f,0.f,0.f,0.f};
  v16h a = loadA(A, lda, m0, 0);
  v16h b = loadB(Bt, ldb, n0, 0);
  for (int k = 32; k < K; k += 32) {
    v16h a2 = loadA(A, lda, m0, k);
    v16h b2 = loadB(Bt, ldb, n0, k);
    acc = __builtin_amdgcn_wmma_f32_16x16x32_f16(false, a, false, b, (short)0, acc, false, false);
    a = a2; b = b2;
  }
  return __builtin_amdgcn_wmma_f32_16x16x32_f16(false, a, false, b, (short)0, acc, false, false);
}

// C layout: N = n0+(lane&15), M = m0 + 8*(lane>>4) + j
__device__ __forceinline__ void gemm_ld(const u16* A, int lda, const u16* Bt, int ldb,
                                        u16* C, int ldc, int coff, const float* bias,
                                        int M, int N, int K, bool relu) {
  int wave = (int)(threadIdx.x >> 5), lane = (int)(threadIdx.x & 31u);
  int nt = N >> 4, jobs = (M >> 4) * nt;
  for (int t = wave; t < jobs; t += NWAVE) {
    int m0 = (t / nt) << 4, n0 = (t % nt) << 4;
    v8f acc = kloop(A, lda, Bt, ldb, K, m0, n0);
    int n = n0 + (lane & 15), mb = m0 + ((lane >> 4) << 3);
    float bv = bias ? bias[n] : 0.f;
#pragma unroll
    for (int j = 0; j < 8; ++j) {
      float v = acc[j] + bv;
      if (relu) v = fmaxf(v, 0.f);
      C[(size_t)(mb + j) * ldc + coff + n] = f2h(v);
    }
  }
}

// Transposed store: Ct[n][m] (row stride ldt) — used to build V^T for attn*V GEMMs.
__device__ __forceinline__ void gemm_tr(const u16* A, int lda, const u16* Bt, int ldb,
                                        u16* Ct, int ldt, const float* bias,
                                        int M, int N, int K) {
  int wave = (int)(threadIdx.x >> 5), lane = (int)(threadIdx.x & 31u);
  int nt = N >> 4, jobs = (M >> 4) * nt;
  for (int t = wave; t < jobs; t += NWAVE) {
    int m0 = (t / nt) << 4, n0 = (t % nt) << 4;
    v8f acc = kloop(A, lda, Bt, ldb, K, m0, n0);
    int n = n0 + (lane & 15), mb = m0 + ((lane >> 4) << 3);
    float bv = bias ? bias[n] : 0.f;
#pragma unroll
    for (int j = 0; j < 8; ++j)
      Ct[(size_t)n * ldt + mb + j] = f2h(acc[j] + bv);
  }
}

// Attention score GEMM: scale, optional per-(item,head) bias, zero cross-item blocks.
__device__ __forceinline__ void gemm_score(const u16* A, int lda, const u16* Bt, int ldb,
                                           u16* C, int ldc, int M, int N, int K,
                                           int rowsA, int colsB, float scale,
                                           const float* bt, int h) {
  int wave = (int)(threadIdx.x >> 5), lane = (int)(threadIdx.x & 31u);
  int nt = N >> 4, jobs = (M >> 4) * nt;
  for (int t = wave; t < jobs; t += NWAVE) {
    int m0 = (t / nt) << 4, n0 = (t % nt) << 4;
    v8f acc = kloop(A, lda, Bt, ldb, K, m0, n0);
    int n = n0 + (lane & 15), mb = m0 + ((lane >> 4) << 3);
#pragma unroll
    for (int j = 0; j < 8; ++j) {
      int m = mb + j;
      u16 out = 0;
      if ((m / rowsA) == (n / colsB)) {                 // same batch item
        float v = acc[j] * scale;
        if (bt) v += bt[((m / rowsA) * 2 + h) * 9 + (m % rowsA) * 3 + (n % colsB)];
        out = f2h(v);
      }
      C[(size_t)m * ldc + n] = out;
    }
  }
}

// ================= K1: embedding MLPs (9x) + interaction-mask MLP =================
struct EmbW { const float *W0,*b0,*W1,*b1,*W2,*b2; };
struct K1Args {
  const float* tau; const float* decay;
  EmbW e[9];                                   // 0..2 tau_emb, 3..8 decay_emb
  const float *mW0,*mb0,*mW1,*mb1,*mW2,*mb2,*mW3,*mb3;
  u16* X;                                      // [B][9][128] f16
  float* biasb;                                // [B][2][3][3] f32
  int ntiles;
};

__global__ __launch_bounds__(BDIM) void tau_k1_emb_mask(K1Args P) {
  extern __shared__ char smem[];
  const int tid   = (int)threadIdx.x;
  const int stage = (int)blockIdx.y;
  if (stage < 9) {
    u16* Wt2   = (u16*)(smem + 0);        // 128 x 512 (transposed W1)
    u16* Wt3   = (u16*)(smem + 131072);   // 128 x 128 (transposed W2)
    u16* Wt0p  = (u16*)(smem + 163840);   // 512 x 32 (transposed W0, K padded 3->32)
    u16* h1    = (u16*)(smem + 196608);   // 16 x 512
    u16* h2    = (u16*)(smem + 212992);   // 16 x 128
    float* b0f = (float*)(smem + 217088);
    float* b2f = (float*)(smem + 219136);
    float* b3f = (float*)(smem + 219648);
    u16* A0    = (u16*)(smem + 220160);   // 16 x 32 (K padded)
    EmbW w = P.e[stage];
    for (int i = tid; i < 65536; i += BDIM) { int k=i>>7, n=i&127; Wt2[n*512+k] = f2h(w.W1[i]); }
    for (int i = tid; i < 16384; i += BDIM) { int k=i>>7, n=i&127; Wt3[n*128+k] = f2h(w.W2[i]); }
    for (int i = tid; i < 16384; i += BDIM) { int n=i>>5, k=i&31; Wt0p[i] = (k<3) ? f2h(w.W0[k*512+n]) : (u16)0; }
    for (int i = tid; i < 512;  i += BDIM) b0f[i] = w.b0[i];
    for (int i = tid; i < 128;  i += BDIM) { b2f[i] = w.b1[i]; b3f[i] = w.b2[i]; }
    for (int i = tid; i < 512;  i += BDIM) A0[i] = 0;   // pads stay zero forever
    __syncthreads();
    const float* src = (stage < 3) ? P.tau : P.decay;
    const int nrow = (stage < 3) ? 3 : 6;
    const int ridx = (stage < 3) ? stage : stage - 3;
    for (int t = (int)blockIdx.x; t < P.ntiles; t += (int)gridDim.x) {
      int i0 = t * TILE;
      if (tid < 48)
        A0[(tid/3)*32 + (tid%3)] = f2h(src[(size_t)(i0 + tid/3) * (nrow*3) + ridx*3 + (tid%3)]);
      __syncthreads();
      gemm_ld(A0, 32, Wt0p, 32, h1, 512, 0, b0f, 16, 512, 32, true);   // layer 1 (K padded)
      __syncthreads();
      gemm_ld(h1, 512, Wt2, 512, h2, 128, 0, b2f, 16, 128, 512, true); // layer 2
      __syncthreads();
      gemm_ld(h2, 128, Wt3, 128, P.X + (size_t)i0*1152 + stage*128, 1152, 0,
              b3f, 16, 128, 128, false);                               // layer 3 -> global
      __syncthreads();
    }
  } else {
    // interaction-mask MLP over 144 pair-rows per tile; only channels 0..1 needed
    u16* Wt1   = (u16*)(smem + 0);        // 64 x 64
    u16* Wt2m  = (u16*)(smem + 8192);     // 64 x 64
    u16* Wt0p  = (u16*)(smem + 16384);    // 64 x 32 (K padded 4->32)
    u16* Af    = (u16*)(smem + 20480);    // 144 x 32 (K padded)
    u16* hA    = (u16*)(smem + 29696);    // 144 x 64
    u16* hB    = (u16*)(smem + 48128);    // 144 x 64
    float* b0f = (float*)(smem + 66560);
    float* b1f = (float*)(smem + 66816);
    float* b2f = (float*)(smem + 67072);
    float* w3f = (float*)(smem + 67328);  // 64 x 2 (sliced last layer)
    float* b3f = (float*)(smem + 67840);
    for (int i = tid; i < 4096; i += BDIM) { int k=i>>6, n=i&63; Wt1[n*64+k]=f2h(P.mW1[i]); Wt2m[n*64+k]=f2h(P.mW2[i]); }
    for (int i = tid; i < 2048; i += BDIM) { int n=i>>5, k=i&31; Wt0p[i] = (k<4) ? f2h(P.mW0[k*64+n]) : (u16)0; }
    for (int i = tid; i < 64; i += BDIM) {
      b0f[i]=P.mb0[i]; b1f[i]=P.mb1[i]; b2f[i]=P.mb2[i];
      w3f[i*2]=P.mW3[i*8]; w3f[i*2+1]=P.mW3[i*8+1];
    }
    if (tid < 2) b3f[tid] = P.mb3[tid];
    for (int i = tid; i < 4608; i += BDIM) Af[i] = 0;
    __syncthreads();
    const float PI = 3.14159265358979323846f, TP = 6.283185307179586477f;
    for (int t = (int)blockIdx.x; t < P.ntiles; t += (int)gridDim.x) {
      int i0 = t * TILE;
      for (int p = tid; p < 144; p += BDIM) {
        int m = p / 9, pr = p % 9, i = pr / 3, j = pr % 3;
        const float* tm = P.tau + (size_t)(i0+m)*9;
        float pti = __expf(tm[i*3]), ptj = __expf(tm[j*3]);
        float dphi = tm[i*3+1] - tm[j*3+1];
        float deta = tm[i*3+2] - tm[j*3+2];
        float r = fmodf(dphi + PI, TP); if (r < 0.f) r += TP; dphi = r - PI;
        float dR = sqrtf(deta*deta + dphi*dphi + 1e-8f);
        float mnp = fminf(pti, ptj);
        float kT = mnp * dR;
        float z  = mnp / (pti + ptj + 1e-8f);
        float m2 = fmaxf(2.f*pti*ptj*(coshf(deta) - cosf(dphi)), 1e-8f);
        Af[p*32+0] = f2h(n2n(logf(dR)));
        Af[p*32+1] = f2h(n2n(logf(kT)));
        Af[p*32+2] = f2h(n2n(logf(z)));
        Af[p*32+3] = f2h(n2n(logf(m2)));
      }
      __syncthreads();
      gemm_ld(Af, 32, Wt0p, 32, hA, 64, 0, b0f, 144, 64, 32, true);  // layer 1 (K padded)
      __syncthreads();
      gemm_ld(hA, 64, Wt1, 64, hB, 64, 0, b1f, 144, 64, 64, true);
      __syncthreads();
      gemm_ld(hB, 64, Wt2m, 64, hA, 64, 0, b2f, 144, 64, 64, true);
      __syncthreads();
      for (int i = tid; i < 288; i += BDIM) {      // sliced last layer (2 channels)
        int m = i >> 1, c = i & 1;
        float v = b3f[c];
        for (int k = 0; k < 64; ++k) v += h2f(hA[m*64+k]) * w3f[k*2+c];
        P.biasb[((size_t)(i0 + m/9)*2 + c)*9 + (m%9)] = v;   // bias[b][h][q][k]=mask[b][q][k][h]
      }
      __syncthreads();
    }
  }
}

// ================= K2: MHSA (stage 0 = tau S=3 w/ bias, stage 1 = prod S=6) =======
struct MhaW { const float *WqW,*Wqb,*WkW,*Wkb,*WvW,*Wvb,*WoW,*Wob; };
struct K2Args { const u16* X; const float* biasb; MhaW mh[2];
                u16* Ytau; u16* Yprod; int ntiles; };

__global__ __launch_bounds__(BDIM) void tau_k2_mhsa(K2Args P) {
  extern __shared__ char smem[];
  const int tid   = (int)threadIdx.x;
  const int stage = (int)blockIdx.y;
  u16* Wtq = (u16*)(smem);
  u16* Wtk = (u16*)(smem + 32768);
  u16* Wtv = (u16*)(smem + 65536);
  u16* Wto = (u16*)(smem + 98304);
  float* bq = (float*)(smem + 131072);
  float* bk = (float*)(smem + 131584);
  float* bv = (float*)(smem + 132096);
  float* bo = (float*)(smem + 132608);
  u16* Xs    = (u16*)(smem + 133120);  // 96 x 128
  u16* Qs    = (u16*)(smem + 157696);  // 96 x 128
  u16* Ks    = (u16*)(smem + 182272);  // 96 x 128
  u16* Vt    = (u16*)(smem + 206848);  // 128 x Rp (V transposed, K padded for S=3)
  u16* Os    = (u16*)(smem + 231424);  // 96 x 128
  u16* attnS = (u16*)(smem + 256000);  // R x Rp scores/attn (block-diagonal)
  float* btile = (float*)(smem + 274432); // 16 x 2 x 9
  MhaW w = P.mh[stage];
  const int S = stage ? 6 : 3;
  const int R = TILE * S;
  const int Rp = stage ? 96 : 64;      // K padded to multiple of 32
  const int xoff = stage ? 3 : 0;
  u16* Y = stage ? P.Yprod : P.Ytau;
  for (int i = tid; i < 16384; i += BDIM) {
    int k = i >> 7, n = i & 127, d = n*128 + k;
    Wtq[d] = f2h(w.WqW[i]); Wtk[d] = f2h(w.WkW[i]);
    Wtv[d] = f2h(w.WvW[i]); Wto[d] = f2h(w.WoW[i]);
  }
  for (int i = tid; i < 128; i += BDIM) { bq[i]=w.Wqb[i]; bk[i]=w.Wkb[i]; bv[i]=w.Wvb[i]; bo[i]=w.Wob[i]; }
  for (int i = tid; i < 9216;  i += BDIM) attnS[i] = 0;  // pads stay zero forever
  for (int i = tid; i < 12288; i += BDIM) Vt[i] = 0;
  __syncthreads();
  for (int t = (int)blockIdx.x; t < P.ntiles; t += (int)gridDim.x) {
    int i0 = t * TILE;
    for (int i = tid; i < R*64; i += BDIM) {
      int r = i >> 6, cu = i & 63;
      ((u32*)(Xs + r*128))[cu] =
        ((const u32*)(P.X + ((size_t)(i0 + r/S)*9 + xoff + r%S)*128))[cu];
    }
    if (stage == 0)
      for (int i = tid; i < 288; i += BDIM) btile[i] = P.biasb[(size_t)i0*18 + i];
    __syncthreads();
    gemm_ld(Xs, 128, Wtq, 128, Qs, 128, 0, bq, R, 128, 128, false);
    gemm_ld(Xs, 128, Wtk, 128, Ks, 128, 0, bk, R, 128, 128, false);
    gemm_tr(Xs, 128, Wtv, 128, Vt, Rp, bv, R, 128, 128);       // V^T
    __syncthreads();
    for (int h = 0; h < 2; ++h) {
      int hb = h * 64;
      // scores = (Q_h)(K_h)^T / 8 + bias, cross-item entries zeroed
      gemm_score(Qs + hb, 128, Ks + hb, 128, attnS, Rp, R, R, 64,
                 S, S, 0.125f, (stage == 0) ? btile : nullptr, h);
      __syncthreads();
      for (int r = tid; r < R; r += BDIM) {                    // per-row softmax
        int item = r / S;
        u16* row = attnS + (size_t)r * Rp + item * S;
        float e[6], mx = -1e30f;
        for (int kk = 0; kk < S; ++kk) { e[kk] = h2f(row[kk]); mx = fmaxf(mx, e[kk]); }
        float sm = 0.f;
        for (int kk = 0; kk < S; ++kk) { e[kk] = __expf(e[kk] - mx); sm += e[kk]; }
        float inv = 1.f / sm;
        for (int kk = 0; kk < S; ++kk) row[kk] = f2h(e[kk] * inv);
      }
      __syncthreads();
      // out_h = attn * V_h  (block-diagonal attn, zero elsewhere)
      gemm_ld(attnS, Rp, Vt + hb*Rp, Rp, Os, 128, hb, nullptr, R, 64, Rp, false);
      __syncthreads();
    }
    gemm_ld(Os, 128, Wto, 128, Y + (size_t)i0 * S * 128, 128, 0, bo, R, 128, 128, false);
    __syncthreads();
  }
}

// ================= K3: cross-attention + fc + regression head =====================
struct K3Args {
  const u16 *Ytau, *Yprod;
  const float *cWq,*cWk,*cWv,*cfcW,*cfcb;
  const float *rW0,*rb0,*rW1,*rb1,*rW2,*rb2,*rW3,*rb3,*rW4,*rb4;
  float* outp; int ntiles;
};

__global__ __launch_bounds__(BDIM) void tau_k3_cross_reg(K3Args P) {
  extern __shared__ char smem[];
  const int tid = (int)threadIdx.x;
  u16* WtQ  = (u16*)(smem);
  u16* WtK  = (u16*)(smem + 32768);
  u16* WtV  = (u16*)(smem + 65536);
  u16* WtF  = (u16*)(smem + 98304);
  float* fcb = (float*)(smem + 131072);
  u16* rWt0 = (u16*)(smem + 131584);   // 64 x 384
  u16* rWt1 = (u16*)(smem + 180736);
  u16* rWt2 = (u16*)(smem + 188928);
  u16* rWt3 = (u16*)(smem + 197120);
  float* rW4f = (float*)(smem + 205312);
  float* rb0 = (float*)(smem + 205824);
  float* rb1 = (float*)(smem + 206080);
  float* rb2 = (float*)(smem + 206336);
  float* rb3 = (float*)(smem + 206592);
  float* rb4 = (float*)(smem + 206848);
  u16* Xt = (u16*)(smem + 206856);     // 48 x 128; later aliased as attnS (48 x 96)
  u16* Yp = (u16*)(smem + 219144);     // 96 x 128
  u16* Qc = (u16*)(smem + 243720);     // 48 x 128; later attention output
  u16* Kc = (u16*)(smem + 256008);     // 96 x 128
  u16* Vt = (u16*)(smem + 280584);     // 128 x 96 (V transposed)
  u16* Fc = (u16*)(smem + 305160);     // 48 x 128 == 16 x 384
  u16* rh1 = (u16*)(smem + 317448);    // 16 x 64
  u16* rh2 = (u16*)(smem + 319496);    // 16 x 64
  for (int i = tid; i < 16384; i += BDIM) {
    int k = i >> 7, n = i & 127, d = n*128 + k;
    WtQ[d] = f2h(P.cWq[i]); WtK[d] = f2h(P.cWk[i]);
    WtV[d] = f2h(P.cWv[i]); WtF[d] = f2h(P.cfcW[i]);
  }
  for (int i = tid; i < 24576; i += BDIM) { int k=i/64, n=i%64; rWt0[n*384+k] = f2h(P.rW0[i]); }
  for (int i = tid; i < 4096; i += BDIM) {
    int k=i>>6, n=i&63;
    rWt1[n*64+k]=f2h(P.rW1[i]); rWt2[n*64+k]=f2h(P.rW2[i]); rWt3[n*64+k]=f2h(P.rW3[i]);
  }
  for (int i = tid; i < 128; i += BDIM) { fcb[i] = P.cfcb[i]; rW4f[i] = P.rW4[i]; }
  for (int i = tid; i < 64; i += BDIM) { rb0[i]=P.rb0[i]; rb1[i]=P.rb1[i]; rb2[i]=P.rb2[i]; rb3[i]=P.rb3[i]; }
  if (tid < 2) rb4[tid] = P.rb4[tid];
  __syncthreads();
  for (int t = (int)blockIdx.x; t < P.ntiles; t += (int)gridDim.x) {
    int i0 = t * TILE;
    for (int i = tid; i < 48*64; i += BDIM) {
      int r = i >> 6, cu = i & 63;
      ((u32*)(Xt + r*128))[cu] = ((const u32*)(P.Ytau + ((size_t)i0*3 + r)*128))[cu];
    }
    for (int i = tid; i < 96*64; i += BDIM) {
      int r = i >> 6, cu = i & 63;
      ((u32*)(Yp + r*128))[cu] = ((const u32*)(P.Yprod + ((size_t)i0*6 + r)*128))[cu];
    }
    __syncthreads();
    gemm_ld(Xt, 128, WtQ, 128, Qc, 128, 0, nullptr, 48, 128, 128, false);
    gemm_ld(Yp, 128, WtK, 128, Kc, 128, 0, nullptr, 96, 128, 128, false);
    gemm_tr(Yp, 128, WtV, 128, Vt, 96, nullptr, 96, 128, 128);   // V^T (128 x 96)
    __syncthreads();
    // energies over full 128 dims; 3x6 blocks kept, rest zeroed; attnS aliases Xt
    gemm_score(Qc, 128, Kc, 128, Xt, 96, 48, 96, 128,
               3, 6, 0.08838834764831845f, nullptr, 0);
    __syncthreads();
    for (int r = tid; r < 48; r += BDIM) {                       // softmax over 6 keys
      u16* row = Xt + (size_t)r * 96 + (r / 3) * 6;
      float e[6], mx = -1e30f;
      for (int kk = 0; kk < 6; ++kk) { e[kk] = h2f(row[kk]); mx = fmaxf(mx, e[kk]); }
      float sm = 0.f;
      for (int kk = 0; kk < 6; ++kk) { e[kk] = __expf(e[kk] - mx); sm += e[kk]; }
      float inv = 1.f / sm;
      for (int kk = 0; kk < 6; ++kk) row[kk] = f2h(e[kk] * inv);
    }
    __syncthreads();
    gemm_ld(Xt, 96, Vt, 96, Qc, 128, 0, nullptr, 48, 128, 96, false); // attn * V -> Qc
    __syncthreads();
    gemm_ld(Qc, 128, WtF, 128, Fc, 128, 0, fcb, 48, 128, 128, false); // fc; Fc == 16x384
    __syncthreads();
    gemm_ld(Fc,  384, rWt0, 384, rh1, 64, 0, rb0, 16, 64, 384, true);
    __syncthreads();
    gemm_ld(rh1, 64, rWt1, 64, rh2, 64, 0, rb1, 16, 64, 64, true);
    __syncthreads();
    gemm_ld(rh2, 64, rWt2, 64, rh1, 64, 0, rb2, 16, 64, 64, true);
    __syncthreads();
    gemm_ld(rh1, 64, rWt3, 64, rh2, 64, 0, rb3, 16, 64, 64, true);
    __syncthreads();
    for (int i = tid; i < 32; i += BDIM) {           // final 64 -> 2 in VALU
      int m = i >> 1, c = i & 1;
      float v = rb4[c];
      for (int k = 0; k < 64; ++k) v += h2f(rh2[m*64 + k]) * rW4f[k*2 + c];
      P.outp[(size_t)(i0 + m)*2 + c] = v;
    }
    __syncthreads();
  }
}

// ================================== host ==========================================
extern "C" void kernel_launch(void* const* d_in, const int* in_sizes, int n_in,
                              void* d_out, int out_size, void* d_ws, size_t ws_size,
                              hipStream_t stream) {
  (void)ws_size; (void)out_size;
  if (n_in < 95) return;
  auto f = [&](int i) { return (const float*)d_in[i]; };
  const int Btot   = in_sizes[0] / 9;
  const int ntiles = Btot / TILE;

  char* ws = (char*)d_ws;
  size_t off = 0;
  auto carve = [&](size_t bytes) { void* p = ws + off; off = (off + bytes + 255) & ~(size_t)255; return p; };
  u16*   X     = (u16*)  carve((size_t)Btot * 1152 * 2);  // [B][9][128] f16
  float* biasb = (float*)carve((size_t)Btot * 18 * 4);    // [B][2][3][3]
  u16*   Ytau  = (u16*)  carve((size_t)Btot * 384 * 2);   // [B][3][128] f16
  u16*   Yprod = (u16*)  carve((size_t)Btot * 768 * 2);   // [B][6][128] f16

  // pytree (alphabetical dict) flatten order:
  // 0 tau, 1 decay, 2..6 cross{Wk,Wq,Wv,fc.W,fc.b}, 7..42 decay_emb,
  // 43..50 mask_emb, 51..58 mha_add{Wk,Wo,Wq,Wv each W,b}, 59..66 mha_tau,
  // 67..76 reg, 77..94 tau_emb
  K1Args a1{};
  a1.tau = f(0); a1.decay = f(1);
  for (int i = 0; i < 3; ++i) { int b = 77 + 6*i; a1.e[i]   = EmbW{f(b),f(b+1),f(b+2),f(b+3),f(b+4),f(b+5)}; }
  for (int i = 0; i < 6; ++i) { int b = 7  + 6*i; a1.e[3+i] = EmbW{f(b),f(b+1),f(b+2),f(b+3),f(b+4),f(b+5)}; }
  a1.mW0=f(43); a1.mb0=f(44); a1.mW1=f(45); a1.mb1=f(46);
  a1.mW2=f(47); a1.mb2=f(48); a1.mW3=f(49); a1.mb3=f(50);
  a1.X = X; a1.biasb = biasb; a1.ntiles = ntiles;

  K2Args a2{};
  a2.X = X; a2.biasb = biasb;
  a2.mh[0] = MhaW{f(63),f(64),f(59),f(60),f(65),f(66),f(61),f(62)}; // mha_tau
  a2.mh[1] = MhaW{f(55),f(56),f(51),f(52),f(57),f(58),f(53),f(54)}; // mha_add
  a2.Ytau = Ytau; a2.Yprod = Yprod; a2.ntiles = ntiles;

  K3Args a3{};
  a3.Ytau = Ytau; a3.Yprod = Yprod;
  a3.cWk = f(2); a3.cWq = f(3); a3.cWv = f(4); a3.cfcW = f(5); a3.cfcb = f(6);
  a3.rW0=f(67); a3.rb0=f(68); a3.rW1=f(69); a3.rb1=f(70); a3.rW2=f(71); a3.rb2=f(72);
  a3.rW3=f(73); a3.rb3=f(74); a3.rW4=f(75); a3.rb4=f(76);
  a3.outp = (float*)d_out; a3.ntiles = ntiles;

  tau_k1_emb_mask <<<dim3(128, 10), BDIM, 221184, stream>>>(a1);
  tau_k2_mhsa     <<<dim3(256, 2),  BDIM, 275584, stream>>>(a2);
  tau_k3_cross_reg<<<dim3(512, 1),  BDIM, 321544, stream>>>(a3);
}
